// SomDST_53669911331268
// MI455X (gfx1250) — matrix operand order repaired and checked by reading, not compiled, not run on hardware
//
#include <hip/hip_runtime.h>
#include <hip/hip_bf16.h>
#include <math.h>

// ---- problem constants ----
#define BB 32
#define SS 512
#define HH 768
#define VV 30522
#define NUPD 3
#define MAXL 8
#define KDIM 768
#define G3H 2304   // 3*H

typedef __attribute__((ext_vector_type(16))) __bf16 v16bf;
typedef __attribute__((ext_vector_type(8)))  float  v8f;

// ---------------------------------------------------------------------------
// WMMA fragment loaders (fp32 memory -> bf16 fragments, contiguous k = 16*grp + s order in A and B)
// A (16x32, 16-bit): lane m=lane&15; grp=lane>>4 selects K-halves:
//   regs 0..7  : K = kb + grp*16 + 0..7
//   regs 8..15 : K = kb + grp*16 + 8 + 0..7
// ---------------------------------------------------------------------------
__device__ inline v16bf load_fragA(const float* __restrict__ A, int row, int kb, int grp) {
    v16bf f;
    const float* p = A + (size_t)row * KDIM + kb + grp * 16;
#pragma unroll
    for (int i = 0; i < 8; ++i) f[i] = (__bf16)p[i];
    p += 8;
#pragma unroll
    for (int i = 0; i < 8; ++i) f[8 + i] = (__bf16)p[i];
    return f;
}

// B (32x16, 16-bit): lane n=lane&15; grp selects K range:
//   regs 0..15 : K = kb + grp*16 + 0..15, column n
// B[k][n] = W[n][k] (W row-major [N x 768]) -> 16 contiguous floats of row n.
__device__ inline v16bf load_fragB(const float* __restrict__ W, int wrow, int kb, int grp) {
    v16bf f;
    const float* p = W + (size_t)wrow * KDIM + kb + grp * 16;
#pragma unroll
    for (int i = 0; i < 16; ++i) f[i] = (__bf16)p[i];
    return f;
}

// ---------------------------------------------------------------------------
// Out[32,N] = A[32,768] @ W[N,768]^T (+ bias).  One wave per 16-col tile,
// both 16-row tiles kept in-wave so each B fragment is loaded once.
// B streams from the 192MB L2 (embed table + GRU weights are L2-resident);
// global_prefetch_b8 pulls the next k-block's W line into the WGP cache.
// ---------------------------------------------------------------------------
__global__ void gemm32_wmma(const float* __restrict__ A,
                            const float* __restrict__ W,
                            const float* __restrict__ bias,
                            float* __restrict__ Out,
                            int N) {
    const int wid  = threadIdx.x >> 5;
    const int lane = threadIdx.x & 31;
    const int tile = blockIdx.x * (blockDim.x >> 5) + wid;
    const int ntiles = (N + 15) >> 4;
    if (tile >= ntiles) return;                 // wave-uniform exit

    const int n0  = tile << 4;
    const int c   = lane & 15;
    const int grp = lane >> 4;
    int wrow = n0 + c;
    if (wrow >= N) wrow = N - 1;                // clamp reads; stores masked

    // warm the WGP cache for this wave's W row and A rows (no counter cost)
    __builtin_prefetch(W + (size_t)wrow * KDIM, 0, 3);
    __builtin_prefetch(A + (size_t)c * KDIM, 0, 3);
    __builtin_prefetch(A + (size_t)(16 + c) * KDIM, 0, 3);

    v8f acc0 = {};
    v8f acc1 = {};
#pragma unroll 4
    for (int kb = 0; kb < KDIM; kb += 32) {
        // prefetch next k-block of the streamed B operand
        if (kb + 32 < KDIM)
            __builtin_prefetch(W + (size_t)wrow * KDIM + kb + 32 + grp * 16, 0, 3);
        v16bf fb  = load_fragB(W, wrow, kb, grp);
        v16bf fa0 = load_fragA(A, c,      kb, grp);
        v16bf fa1 = load_fragA(A, 16 + c, kb, grp);
        acc0 = __builtin_amdgcn_wmma_f32_16x16x32_bf16(false, fa0, false, fb,
                                                       (short)0, acc0, false, false);
        acc1 = __builtin_amdgcn_wmma_f32_16x16x32_bf16(false, fa1, false, fb,
                                                       (short)0, acc1, false, false);
    }

    if (n0 + c < N) {
        const float bv = bias ? bias[n0 + c] : 0.0f;
#pragma unroll
        for (int r = 0; r < 8; ++r) {
            const int m = r + 8 * grp;          // C/D layout: vgpr r, group grp
            Out[(size_t)m        * N + n0 + c] = acc0[r] + bv;
            Out[(size_t)(16 + m) * N + n0 + c] = acc1[r] + bv;
        }
    }
}

// ---------------------------------------------------------------------------
// Step input: slot j step k -> decoder_input[:,j] (k==0) else embed(teacher)
// ---------------------------------------------------------------------------
__global__ void build_w_kernel(const float* __restrict__ dec_in,
                               const int* __restrict__ teacher,
                               const float* __restrict__ embed,
                               float* __restrict__ w, int j, int k) {
    int idx = blockIdx.x * blockDim.x + threadIdx.x;
    if (idx >= BB * HH) return;
    int b = idx / HH, hh = idx - b * HH;
    float val;
    if (k == 0) {
        val = dec_in[((size_t)b * NUPD + j) * HH + hh];
    } else {
        int tok = teacher[((size_t)b * NUPD + j) * MAXL + (k - 1)];
        val = embed[(size_t)tok * HH + hh];
    }
    w[idx] = val;
}

// ---------------------------------------------------------------------------
// Fused GRU gates: h = (1-z)*n + z*h  (in-place, one thread per element)
// ---------------------------------------------------------------------------
__global__ void gru_gates_kernel(const float* __restrict__ gi,
                                 const float* __restrict__ gh,
                                 float* __restrict__ h) {
    int idx = blockIdx.x * blockDim.x + threadIdx.x;
    if (idx >= BB * HH) return;
    int b = idx / HH, cd = idx - b * HH;
    const float* gib = gi + (size_t)b * G3H;
    const float* ghb = gh + (size_t)b * G3H;
    float ir = gib[cd], iz = gib[HH + cd], in_ = gib[2 * HH + cd];
    float hr = ghb[cd], hz = ghb[HH + cd], hn  = ghb[2 * HH + cd];
    float r = 1.0f / (1.0f + __expf(-(ir + hr)));
    float z = 1.0f / (1.0f + __expf(-(iz + hz)));
    float n = tanhf(in_ + r * hn);
    h[idx] = (1.0f - z) * n + z * h[idx];
}

// ---------------------------------------------------------------------------
// Encoder attention + context + p_gen: one workgroup (256 thr) per batch row
// ---------------------------------------------------------------------------
__global__ void attention_kernel(const float* __restrict__ enc,
                                 const int* __restrict__ x,
                                 const float* __restrict__ h,
                                 const float* __restrict__ w,
                                 const float* __restrict__ wgen_w,
                                 const float* __restrict__ wgen_b,
                                 float* __restrict__ attn_hist,
                                 float* __restrict__ pgen) {
    __shared__ float sh[HH];
    __shared__ float se[SS];
    __shared__ float red[256];
    const int b = blockIdx.x, t = threadIdx.x;

    for (int i = t; i < HH; i += 256) sh[i] = h[(size_t)b * HH + i];
    __syncthreads();

    for (int s = t; s < SS; s += 256) {
        const float* ep = enc + ((size_t)b * SS + s) * HH;
        float acc = 0.0f;
#pragma unroll 4
        for (int i = 0; i < HH; ++i) acc = fmaf(ep[i], sh[i], acc);
        se[s] = (x[(size_t)b * SS + s] == 0) ? -1e9f : acc;
    }
    __syncthreads();

    float m = -INFINITY;
    for (int s = t; s < SS; s += 256) m = fmaxf(m, se[s]);
    red[t] = m; __syncthreads();
    for (int o = 128; o > 0; o >>= 1) { if (t < o) red[t] = fmaxf(red[t], red[t + o]); __syncthreads(); }
    m = red[0]; __syncthreads();

    float sum = 0.0f;
    for (int s = t; s < SS; s += 256) { float e = __expf(se[s] - m); se[s] = e; sum += e; }
    red[t] = sum; __syncthreads();
    for (int o = 128; o > 0; o >>= 1) { if (t < o) red[t] += red[t + o]; __syncthreads(); }
    sum = red[0]; __syncthreads();

    const float inv = 1.0f / sum;
    for (int s = t; s < SS; s += 256) { se[s] *= inv; attn_hist[(size_t)b * SS + s] = se[s]; }
    __syncthreads();

    // context: thread t owns h-dims t, t+256, t+512 (coalesced across threads)
    float c0 = 0.f, c1 = 0.f, c2 = 0.f;
    for (int s = 0; s < SS; ++s) {
        const float a = se[s];
        const float* ep = enc + ((size_t)b * SS + s) * HH;
        c0 = fmaf(a, ep[t],       c0);
        c1 = fmaf(a, ep[t + 256], c1);
        c2 = fmaf(a, ep[t + 512], c2);
    }
    // p_gen = sigmoid([w, h, ctx] . wgen_w + b)
    const float* wb = w + (size_t)b * HH;
    float part = 0.f;
    part += wb[t] * wgen_w[t] + wb[t + 256] * wgen_w[t + 256] + wb[t + 512] * wgen_w[t + 512];
    part += sh[t] * wgen_w[HH + t] + sh[t + 256] * wgen_w[HH + t + 256] + sh[t + 512] * wgen_w[HH + t + 512];
    part += c0 * wgen_w[2 * HH + t] + c1 * wgen_w[2 * HH + t + 256] + c2 * wgen_w[2 * HH + t + 512];
    red[t] = part; __syncthreads();
    for (int o = 128; o > 0; o >>= 1) { if (t < o) red[t] += red[t + o]; __syncthreads(); }
    if (t == 0) pgen[b] = 1.0f / (1.0f + __expf(-(red[0] + wgen_b[0])));
}

// ---------------------------------------------------------------------------
// Per-row softmax stats over V=30522 logits
// ---------------------------------------------------------------------------
__global__ void vocab_stats_kernel(const float* __restrict__ logits,
                                   float* __restrict__ rowmax,
                                   float* __restrict__ rowsum) {
    __shared__ float red[256];
    const int b = blockIdx.x, t = threadIdx.x;
    const float* lp = logits + (size_t)b * VV;
    float m = -INFINITY;
    for (int v = t; v < VV; v += 256) m = fmaxf(m, lp[v]);
    red[t] = m; __syncthreads();
    for (int o = 128; o > 0; o >>= 1) { if (t < o) red[t] = fmaxf(red[t], red[t + o]); __syncthreads(); }
    m = red[0]; __syncthreads();
    float s = 0.0f;
    for (int v = t; v < VV; v += 256) s += __expf(lp[v] - m);
    red[t] = s; __syncthreads();
    for (int o = 128; o > 0; o >>= 1) { if (t < o) red[t] += red[t + o]; __syncthreads(); }
    if (t == 0) { rowmax[b] = m; rowsum[b] = red[0]; }
}

// ---------------------------------------------------------------------------
// out[b,j,k,:] = p_gen * softmax(logits)   (full overwrite — graph-replay safe)
// ---------------------------------------------------------------------------
__global__ void write_pfinal_kernel(const float* __restrict__ logits,
                                    const float* __restrict__ rowmax,
                                    const float* __restrict__ rowsum,
                                    const float* __restrict__ pgen,
                                    float* __restrict__ out, int j, int k) {
    int idx = blockIdx.x * blockDim.x + threadIdx.x;
    if (idx >= BB * VV) return;
    int b = idx / VV, v = idx - b * VV;
    float val = pgen[b] * __expf(logits[idx] - rowmax[b]) / rowsum[b];
    out[((((size_t)b * NUPD + j) * MAXL + k) * VV) + v] = val;
}

// ---------------------------------------------------------------------------
// out[b,j,k, x[b,s]] += (1-p_gen)*attn_hist[b,s].  One thread per row:
// serial RMW, no float atomics -> bit-deterministic across graph replays.
// ---------------------------------------------------------------------------
__global__ void scatter_ctx_kernel(const int* __restrict__ x,
                                   const float* __restrict__ attn_hist,
                                   const float* __restrict__ pgen,
                                   float* __restrict__ out, int j, int k) {
    int b = blockIdx.x * blockDim.x + threadIdx.x;
    if (b >= BB) return;
    const float coef = 1.0f - pgen[b];
    float* op = out + (((size_t)b * NUPD + j) * MAXL + k) * VV;
    const int* xb = x + (size_t)b * SS;
    const float* ab = attn_hist + (size_t)b * SS;
    for (int s = 0; s < SS; ++s) {
        op[xb[s]] += coef * ab[s];
    }
}

// ---------------------------------------------------------------------------
extern "C" void kernel_launch(void* const* d_in, const int* in_sizes, int n_in,
                              void* d_out, int out_size, void* d_ws, size_t ws_size,
                              hipStream_t stream) {
    (void)in_sizes; (void)n_in; (void)out_size; (void)ws_size;

    const int*   x        = (const int*)  d_in[0];
    const float* dec_in   = (const float*)d_in[1];
    const float* enc      = (const float*)d_in[2];
    const float* hidden   = (const float*)d_in[3];
    const int*   teacher  = (const int*)  d_in[4];
    const float* embed_w  = (const float*)d_in[5];
    const float* gru_w_ih = (const float*)d_in[6];
    const float* gru_w_hh = (const float*)d_in[7];
    const float* gru_b_ih = (const float*)d_in[8];
    const float* gru_b_hh = (const float*)d_in[9];
    const float* wgen_w   = (const float*)d_in[10];
    const float* wgen_b   = (const float*)d_in[11];
    float* out = (float*)d_out;

    // workspace layout (floats)
    float* ws = (float*)d_ws;
    float* h         = ws;                       // 32*768
    float* w         = h  + BB * HH;             // 32*768
    float* gi        = w  + BB * HH;             // 32*2304
    float* gh        = gi + BB * G3H;            // 32*2304
    float* logits    = gh + BB * G3H;            // 32*30522
    float* attn_hist = logits + (size_t)BB * VV; // 32*512
    float* pgen      = attn_hist + BB * SS;      // 32
    float* rowmax    = pgen + BB;                // 32
    float* rowsum    = rowmax + BB;              // 32

    // h <- hidden[0]   (re-init every call: h is mutated across the 24 steps)
    hipMemcpyAsync(h, hidden, (size_t)BB * HH * sizeof(float),
                   hipMemcpyDeviceToDevice, stream);

    const dim3 blk256(256);
    const int elemBH   = (BB * HH + 255) / 256;          // 96 blocks
    const int tilesG   = (G3H + 15) / 16;                // 144 tiles
    const int gridG    = (tilesG + 7) / 8;               // 8 waves/block
    const int tilesV   = (VV + 15) / 16;                 // 1908 tiles
    const int gridV    = (tilesV + 7) / 8;
    const int elemBV   = (BB * VV + 255) / 256;

    for (int t = 0; t < NUPD * MAXL; ++t) {
        const int j = t / MAXL, k = t % MAXL;

        build_w_kernel<<<elemBH, blk256, 0, stream>>>(dec_in, teacher, embed_w, w, j, k);

        // GRU pre-activations (WMMA bf16, fp32 accumulate)
        gemm32_wmma<<<gridG, blk256, 0, stream>>>(w, gru_w_ih, gru_b_ih, gi, G3H);
        gemm32_wmma<<<gridG, blk256, 0, stream>>>(h, gru_w_hh, gru_b_hh, gh, G3H);
        gru_gates_kernel<<<elemBH, blk256, 0, stream>>>(gi, gh, h);

        // vocab logits: dominant GEMM, B streams from the 192MB L2
        gemm32_wmma<<<gridV, blk256, 0, stream>>>(h, embed_w, nullptr, logits, VV);

        attention_kernel<<<BB, blk256, 0, stream>>>(enc, x, h, w, wgen_w, wgen_b,
                                                    attn_hist, pgen);
        vocab_stats_kernel<<<BB, blk256, 0, stream>>>(logits, rowmax, rowsum);
        write_pfinal_kernel<<<elemBV, blk256, 0, stream>>>(logits, rowmax, rowsum,
                                                           pgen, out, j, k);
        scatter_ctx_kernel<<<1, BB, 0, stream>>>(x, attn_hist, pgen, out, j, k);
    }
}
